// Encoder_19327352832539
// MI455X (gfx1250) — matrix operand order repaired and checked by reading, not compiled, not run on hardware
//
#include <hip/hip_runtime.h>
#include <hip/hip_bf16.h>
#include <math.h>

// ---------------------------------------------------------------------------
// Problem dims (match reference)
// ---------------------------------------------------------------------------
#define VOCAB 32000
#define HID   1024
#define SEQ   4096
#define NBLK_RNN 16          // 16 persistent WGs, 64 rows of W_hh each (256KB LDS)
#define ROWS_PER_BLK (HID / NBLK_RNN)   // 64

typedef __bf16 bf16;
typedef bf16  v16bf __attribute__((ext_vector_type(16)));
typedef bf16  v8bf  __attribute__((ext_vector_type(8)));
typedef bf16  v4bf  __attribute__((ext_vector_type(4)));
typedef float v8f   __attribute__((ext_vector_type(8)));

typedef unsigned int u32x4 __attribute__((ext_vector_type(4)));
typedef int          i32x4 __attribute__((ext_vector_type(4)));
typedef int          i32x8 __attribute__((ext_vector_type(8)));

#if defined(__has_builtin)
#if __has_builtin(__builtin_amdgcn_tensor_load_to_lds) && \
    __has_builtin(__builtin_amdgcn_s_wait_tensorcnt)
#define HAVE_TDM 1
#endif
#endif

// Workspace layout (bytes)
#define EMB_OFF 0u                      // bf16 emb [SEQ][HID]      : 8 MiB
#define WHI_OFF (8u << 20)              // bf16 W_hi [HID][HID]     : 2 MiB
#define PRE_OFF (10u << 20)             // f32 pre  [SEQ][HID]      : 16 MiB
#define CTR_OFF (26u << 20)             // u32 ctr  [SEQ]           : 16 KiB

// ---------------------------------------------------------------------------
// 0) Per-call init: zero the per-step barrier counters, write h0 -> out row 0.
// ---------------------------------------------------------------------------
__global__ __launch_bounds__(256) void init_kernel(unsigned* __restrict__ ctr,
                                                   const float* __restrict__ h0,
                                                   float* __restrict__ out) {
    int i = blockIdx.x * 256 + threadIdx.x;
    if (i < SEQ) ctr[i] = 0u;
    int j = i - SEQ;
    if (j >= 0 && j < HID) out[j] = h0[j];
}

// ---------------------------------------------------------------------------
// 1) Embedding gather + L2 normalize (eps=1e-12) -> bf16
// ---------------------------------------------------------------------------
__global__ __launch_bounds__(256) void embed_norm_kernel(const int*   __restrict__ src,
                                                         const float* __restrict__ W,
                                                         bf16*        __restrict__ emb) {
    const int t   = blockIdx.x;
    const int tid = threadIdx.x;
    const float* row = W + (size_t)src[t] * HID;

    float4 v = ((const float4*)row)[tid];
    float s  = v.x * v.x + v.y * v.y + v.z * v.z + v.w * v.w;

    for (int o = 16; o > 0; o >>= 1) s += __shfl_down(s, o, 32);

    __shared__ float warp_s[8];
    if ((tid & 31) == 0) warp_s[tid >> 5] = s;
    __syncthreads();
    float tot = 0.f;
#pragma unroll
    for (int i = 0; i < 8; ++i) tot += warp_s[i];

    const float scale = 1.0f / fmaxf(sqrtf(tot), 1e-12f);
    v4bf o;
    o.x = (bf16)(v.x * scale);
    o.y = (bf16)(v.y * scale);
    o.z = (bf16)(v.z * scale);
    o.w = (bf16)(v.w * scale);
    *(v4bf*)(emb + (size_t)t * HID + tid * 4) = o;
}

// ---------------------------------------------------------------------------
// 2) W_hi f32 -> bf16 (row-major [n][k]; k contiguous = contraction dim)
// ---------------------------------------------------------------------------
__global__ __launch_bounds__(256) void cvt_whi_kernel(const float* __restrict__ whi,
                                                      bf16*        __restrict__ whib) {
    const size_t i = ((size_t)blockIdx.x * 256 + threadIdx.x) * 4;
    float4 v = *(const float4*)(whi + i);
    v4bf o;
    o.x = (bf16)v.x; o.y = (bf16)v.y; o.z = (bf16)v.z; o.w = (bf16)v.w;
    *(v4bf*)(whib + i) = o;
}

// ---------------------------------------------------------------------------
// 3) pre[t][n] = sum_k emb[t][k] * W_hi[n][k]  — bf16 WMMA GEMM
//    Block 256 thr = 8 waves; block tile M64 x N64; wave tile 16x32
//    (A fragment reused across 2 WMMAs). Double-buffered LDS, one barrier
//    per K-step, global loads of the next tile overlap current WMMAs.
// ---------------------------------------------------------------------------
__global__ __launch_bounds__(256) void gemm_whi_kernel(const bf16* __restrict__ emb,
                                                       const bf16* __restrict__ whib,
                                                       float*      __restrict__ pre) {
    __shared__ __align__(32) bf16 As[2][64 * 32];   // [m][k]
    __shared__ __align__(32) bf16 Bs[2][64 * 32];   // [n][k]

    const int tid  = threadIdx.x;
    const int lane = tid & 31;
    const int wid  = tid >> 5;       // 0..7
    const int wm   = wid & 3;        // 4 M-subtiles of 16
    const int wn   = wid >> 2;       // 2 N-subtiles of 32
    const int half = lane >> 4;      // hi/lo 16 lanes
    const int l16  = lane & 15;

    const int n0 = blockIdx.x * 64;  // HID/64 = 16
    const int m0 = blockIdx.y * 64;  // SEQ/64 = 64

    // cooperative global->LDS indices: 256 thr x 16B covers one 4KB tile
    const int r  = tid >> 2;         // 0..63
    const int c8 = (tid & 3) * 8;    // 0,8,16,24

    const bf16* ag = emb  + (size_t)(m0 + r) * HID + c8;
    const bf16* bg = whib + (size_t)(n0 + r) * HID + c8;

    // preload K-tile 0 into buffer 0
    *(v8bf*)&As[0][r * 32 + c8] = *(const v8bf*)(ag);
    *(v8bf*)&Bs[0][r * 32 + c8] = *(const v8bf*)(bg);

    v8f c0 = {}, c1 = {};
    int buf = 0;

    for (int k0 = 0; k0 < HID; k0 += 32) {
        __syncthreads();   // tile[buf] ready; tile[buf^1] free to overwrite
        if (k0 + 32 < HID) {
            *(v8bf*)&As[buf ^ 1][r * 32 + c8] = *(const v8bf*)(ag + k0 + 32);
            *(v8bf*)&Bs[buf ^ 1][r * 32 + c8] = *(const v8bf*)(bg + k0 + 32);
            __builtin_prefetch(ag + k0 + 64, 0, 1);   // global_prefetch_b8
        }

        // A fragment: lane (l16,half): row wm*16+l16,
        // K {half*16..+15} contiguous
        const bf16* arow = &As[buf][(wm * 16 + l16) * 32 + half * 16];
        v8bf a0 = *(const v8bf*)(arow);
        v8bf a1 = *(const v8bf*)(arow + 8);
        v16bf a = __builtin_shufflevector(a0, a1, 0, 1, 2, 3, 4, 5, 6, 7,
                                          8, 9, 10, 11, 12, 13, 14, 15);

        // Two B fragments (cols wn*32..+15 and wn*32+16..+31)
        v16bf b0 = *(const v16bf*)&Bs[buf][(wn * 32 + l16) * 32 + half * 16];
        v16bf b1 = *(const v16bf*)&Bs[buf][(wn * 32 + 16 + l16) * 32 + half * 16];

        c0 = __builtin_amdgcn_wmma_f32_16x16x32_bf16(false, a, false, b0,
                                                     (short)0, c0, false, false);
        c1 = __builtin_amdgcn_wmma_f32_16x16x32_bf16(false, a, false, b1,
                                                     (short)0, c1, false, false);
        buf ^= 1;
    }

    // C layout: VGPR v -> row (half*8 + v), col l16
    const int colb = n0 + wn * 32 + l16;
#pragma unroll
    for (int v = 0; v < 8; ++v) {
        const int row = m0 + wm * 16 + half * 8 + v;
        pre[(size_t)row * HID + colb]      = c0[v];
        pre[(size_t)row * HID + colb + 16] = c1[v];
    }
}

// ---------------------------------------------------------------------------
// 4) Sequential recurrence: 16 persistent WGs, each owns 64 rows of W_hh
//    in LDS (256 KB of the 320 KB WGP LDS), staged via the Tensor Data
//    Mover when available. Per step: DEV-scope load of h_{t-1}, 64x1024
//    matvec slice, tanh, publish, per-step counter barrier.
// ---------------------------------------------------------------------------
__global__ __launch_bounds__(256) void rnn_kernel(const float* __restrict__ whh,
                                                  const float* __restrict__ bias,
                                                  const float* __restrict__ pre,
                                                  float*       __restrict__ out,
                                                  unsigned*    __restrict__ ctr) {
    extern __shared__ float lds[];
    float* Wl = lds;                       // [64][1024] at dynamic-LDS base (offset 0)
    float* hl = lds + ROWS_PER_BLK * HID;  // [1024]

    const int tid   = threadIdx.x;
    const int lane  = tid & 31;
    const int wid   = tid >> 5;
    const int gbase = blockIdx.x * ROWS_PER_BLK;

#if defined(HAVE_TDM)
    // Tensor Data Mover: DMA the 2D tile W_hh[gbase:gbase+64, 0:1024] (f32,
    // row stride 1024) into LDS offset 0. D# per ISA 08_async_tensor §8.
    if (wid == 0) {   // one wave issues; TDM ignores EXEC, tracked by TENSORcnt
        const unsigned long long ga =
            (unsigned long long)(uintptr_t)(whh + (size_t)gbase * HID);
        u32x4 g0;
        g0[0] = 0x1u;                                   // count=1, user mode
        g0[1] = 0u;                                     // lds_addr = 0 (Wl)
        g0[2] = (unsigned)(ga & 0xFFFFFFFFu);           // global_addr[31:0]
        g0[3] = (unsigned)((ga >> 32) & 0x1FFFFFFu)     // global_addr[56:32]
              | (2u << 30);                             // type=2 ("image")
        i32x8 g1;
        g1[0] = (2 << 16);            // data_size=2 -> 4-byte elements
        g1[1] = (int)(HID << 16);     // tensor_dim0[15:0] (=1024) in [31:16]
        g1[2] = (int)(HID << 16);     // tensor_dim0 hi=0 | tensor_dim1 lo=1024
        g1[3] = (int)(HID << 16);     // tensor_dim1 hi=0 | tile_dim0=1024
        g1[4] = ROWS_PER_BLK;         // tile_dim1=64, tile_dim2=0
        g1[5] = HID;                  // tensor_dim0_stride[31:0]=1024
        g1[6] = 0;
        g1[7] = 0;
        i32x4 z4 = {0, 0, 0, 0};
#if defined(__clang_major__) && (__clang_major__ >= 23)
        i32x8 z8 = {0, 0, 0, 0, 0, 0, 0, 0};
        __builtin_amdgcn_tensor_load_to_lds(g0, g1, z4, z4, z8, 0);
#else
        __builtin_amdgcn_tensor_load_to_lds(g0, g1, z4, z4, 0);
#endif
        __builtin_amdgcn_s_wait_tensorcnt(0);           // s_wait_tensorcnt 0
    }
#else
    {
        const float4* srcp = (const float4*)(whh + (size_t)gbase * HID);
        float4*       dstp = (float4*)Wl;
        for (int i = tid; i < ROWS_PER_BLK * HID / 4; i += 256) dstp[i] = srcp[i];
    }
#endif
    __syncthreads();

    for (int t = 0; t < SEQ; ++t) {
        // h_{t-1} -> LDS with device-scope loads (cross-WGP coherent,
        // graph-replay safe: never trusts a stale L0/L1 line).
        const float* hrow = out + (size_t)t * HID;
        for (int i = tid; i < HID; i += 256) {
            hl[i] = __hip_atomic_load(hrow + i, __ATOMIC_RELAXED,
                                      __HIP_MEMORY_SCOPE_AGENT);
        }
        __syncthreads();

        // Each wave computes 8 output rows; lane l handles k = l + 32*i
        // (conflict-free: consecutive lanes -> consecutive banks).
#pragma unroll
        for (int j = 0; j < 8; ++j) {
            const int r = wid * 8 + j;
            const float* wr = Wl + r * HID;
            float acc = 0.f;
#pragma unroll
            for (int i = 0; i < HID / 32; ++i) {
                acc = fmaf(wr[lane + 32 * i], hl[lane + 32 * i], acc);
            }
            for (int o = 16; o > 0; o >>= 1) acc += __shfl_down(acc, o, 32);
            if (lane == 0) {
                const int gr = gbase + r;
                const float z = acc + pre[(size_t)t * HID + gr] + bias[gr];
                out[(size_t)(t + 1) * HID + gr] = tanhf(z);
            }
        }
        __syncthreads();

        // Inter-WG barrier: release our slice, arrive, spin for all 16.
        if (tid == 0) {
            __threadfence();
            __hip_atomic_fetch_add(&ctr[t], 1u, __ATOMIC_RELEASE,
                                   __HIP_MEMORY_SCOPE_AGENT);
            while (__hip_atomic_load(&ctr[t], __ATOMIC_ACQUIRE,
                                     __HIP_MEMORY_SCOPE_AGENT) < (unsigned)NBLK_RNN) {
                __builtin_amdgcn_s_sleep(2);
            }
        }
        // Hardware cluster-barrier fast path (NOP when not dispatched as a
        // cluster; s_barrier_signal/wait -3 on gfx1250).
        __builtin_amdgcn_s_cluster_barrier();
        __syncthreads();
    }
}

// ---------------------------------------------------------------------------
// Launcher
// ---------------------------------------------------------------------------
extern "C" void kernel_launch(void* const* d_in, const int* in_sizes, int n_in,
                              void* d_out, int out_size, void* d_ws, size_t ws_size,
                              hipStream_t stream) {
    const int*   src = (const int*)  d_in[0];
    const float* W   = (const float*)d_in[1];
    const float* h0  = (const float*)d_in[2];
    const float* Whi = (const float*)d_in[3];
    const float* Whh = (const float*)d_in[4];
    const float* b   = (const float*)d_in[5];
    float* out = (float*)d_out;

    char* ws = (char*)d_ws;
    bf16*     emb  = (bf16*)(ws + EMB_OFF);
    bf16*     whib = (bf16*)(ws + WHI_OFF);
    float*    pre  = (float*)(ws + PRE_OFF);
    unsigned* ctr  = (unsigned*)(ws + CTR_OFF);

    // 0) zero barrier counters + seed out row 0 with h0
    init_kernel<<<(SEQ + HID + 255) / 256, 256, 0, stream>>>(ctr, h0, out);

    // 1) gather + L2-normalize -> bf16
    embed_norm_kernel<<<SEQ, 256, 0, stream>>>(src, W, emb);

    // 2) W_hi -> bf16
    cvt_whi_kernel<<<(HID * HID) / (256 * 4), 256, 0, stream>>>(Whi, whib);

    // 3) pre = emb @ W_hi^T via bf16 WMMA (64x64 block tiles)
    dim3 ggrid(HID / 64, SEQ / 64);
    gemm_whi_kernel<<<ggrid, 256, 0, stream>>>(emb, whib, pre);

    // 4) sequential recurrence, 16 persistent WGs, W_hh resident in LDS
    const size_t shmem = (size_t)(ROWS_PER_BLK * HID + HID) * sizeof(float); // 260 KB
    rnn_kernel<<<NBLK_RNN, 256, shmem, stream>>>(Whh, b, pre, out, ctr);
}
